// ProtoGNN_9337258901637
// MI455X (gfx1250) — compile-verified, matrix operand order checked by reading, not run on hardware
//
#include <hip/hip_runtime.h>
#include <math.h>

// ---------------- problem constants ----------------
#define BB 4
#define NN 4096
#define FF 1024
#define DD 512
#define TT 6
#define PP 8
#define HH 8
#define DH 64
#define LL 12
#define NSPLIT 16

typedef __attribute__((ext_vector_type(16))) __bf16 v16bf;
typedef __attribute__((ext_vector_type(8)))  float  v8f;

// =====================================================================
// Kernel 0: zero the small accumulators (cnt, bmsum)
// =====================================================================
__global__ void pg_zero(float* cnt, float* bmsum) {
    if (threadIdx.x < BB * TT) cnt[threadIdx.x] = 0.f;
    if (threadIdx.x < BB) bmsum[threadIdx.x] = 0.f;
}

// =====================================================================
// Kernel 1: validity flags + type counts + batch-mask sums
// one wave (32 lanes) per (b,n) row
// =====================================================================
__global__ __launch_bounds__(256) void pg_valid(const float* __restrict__ x,
                                                const int* __restrict__ bm,
                                                const int* __restrict__ ntype,
                                                float* __restrict__ valid,
                                                float* __restrict__ cnt,
                                                float* __restrict__ bmsum) {
    int wv = threadIdx.x >> 5, lane = threadIdx.x & 31;
    int row = blockIdx.x * 8 + wv;            // < BB*NN
    int b = row >> 12;                        // NN = 4096
    const float* xr = x + (size_t)row * FF;
    int nz = 0;
    for (int f = lane; f < FF; f += 32) nz |= (xr[f] != 0.0f);
    for (int off = 16; off; off >>= 1) nz |= __shfl_xor(nz, off, 32);
    int bmv = bm[row];
    float vld = (bmv != 0 && nz != 0) ? 1.0f : 0.0f;
    if (lane == 0) {
        valid[row] = vld;
        if (vld > 0.f) atomicAdd(cnt + b * TT + ntype[row], 1.0f); // integer-valued: exact
        if (bmv) atomicAdd(bmsum + b, 1.0f);                       // integer-valued: exact
    }
}

// =====================================================================
// Kernel 2: masked per-type partial sums + unmasked fallback partial sums
// grid (FF/256, BB, NSPLIT)
// =====================================================================
__global__ __launch_bounds__(256) void pg_sums(const float* __restrict__ x,
                                               const float* __restrict__ valid,
                                               const int* __restrict__ ntype,
                                               float* __restrict__ sums_part,
                                               float* __restrict__ fall_part) {
    int f = blockIdx.x * 256 + threadIdx.x;
    int b = blockIdx.y, s = blockIdx.z;
    float acc[TT] = {0.f, 0.f, 0.f, 0.f, 0.f, 0.f};
    float fall = 0.f;
    int n0 = s * (NN / NSPLIT);
    for (int n = n0; n < n0 + NN / NSPLIT; ++n) {
        int row = b * NN + n;
        float xv = x[(size_t)row * FF + f];
        fall += xv;
        float vv = (valid[row] > 0.f) ? xv : 0.f;
        int tt = ntype[row];
#pragma unroll
        for (int t = 0; t < TT; ++t) acc[t] += (tt == t) ? vv : 0.f;
    }
#pragma unroll
    for (int t = 0; t < TT; ++t)
        sums_part[((size_t)(s * BB + b) * TT + t) * FF + f] = acc[t];
    fall_part[(size_t)(s * BB + b) * FF + f] = fall;
}

// =====================================================================
// Kernel 3: reduce partials in fixed order (deterministic)
// grid 112: first 96 blocks -> sums (B*T*F), last 16 -> fallback (B*F)
// =====================================================================
__global__ __launch_bounds__(256) void pg_reduce(const float* __restrict__ sums_part,
                                                 const float* __restrict__ fall_part,
                                                 float* __restrict__ sums,
                                                 float* __restrict__ fall) {
    int i = blockIdx.x * 256 + threadIdx.x;
    if (blockIdx.x < 96) {
        float a = 0.f;
        for (int s = 0; s < NSPLIT; ++s) a += sums_part[(size_t)s * BB * TT * FF + i];
        sums[i] = a;
    } else {
        int j = i - 96 * 256;  // < BB*FF
        float a = 0.f;
        for (int s = 0; s < NSPLIT; ++s) a += fall_part[(size_t)s * BB * FF + j];
        fall[j] = a;
    }
}

// =====================================================================
// Kernel 4: proto_mu -> mu = gelu(ln(proto_mu@pm_W+b,g,bn)) -> q0 state
// one block per (b,t); also writes proto_mu output
// =====================================================================
__global__ __launch_bounds__(256) void pg_mu(const float* __restrict__ sums,
                                             const float* __restrict__ cnt,
                                             const float* __restrict__ fall,
                                             const float* __restrict__ bmsum,
                                             const float* __restrict__ pmW,
                                             const float* __restrict__ pmb,
                                             const float* __restrict__ pmg,
                                             const float* __restrict__ pmbn,
                                             const float* __restrict__ pls,
                                             const float* __restrict__ rmul,
                                             float* __restrict__ q,
                                             float* __restrict__ out_pmu) {
    int b = blockIdx.x / TT, t = blockIdx.x % TT;
    __shared__ float pmu[FF];
    __shared__ float y[DD];
    __shared__ float red[2][8];
    float c = cnt[b * TT + t];
    float inv = 1.0f / fmaxf(c, 1.0f);
    float binv = 1.0f / bmsum[b];
    for (int f = threadIdx.x; f < FF; f += 256) {
        float v = (c > 0.f) ? sums[(size_t)(b * TT + t) * FF + f] * inv
                            : fall[(size_t)b * FF + f] * binv;
        pmu[f] = v;
        out_pmu[(size_t)(b * TT + t) * FF + f] = v;
    }
    __syncthreads();
    for (int d = threadIdx.x; d < DD; d += 256) {
        float acc = pmb[d];
        for (int f = 0; f < FF; ++f) acc += pmu[f] * pmW[(size_t)f * DD + d];
        y[d] = acc;
    }
    __syncthreads();
    float ls = 0.f, lss = 0.f;
    for (int d = threadIdx.x; d < DD; d += 256) { float v = y[d]; ls += v; lss += v * v; }
    for (int off = 16; off; off >>= 1) { ls += __shfl_xor(ls, off, 32); lss += __shfl_xor(lss, off, 32); }
    int lane = threadIdx.x & 31, wv = threadIdx.x >> 5;
    if (lane == 0) { red[0][wv] = ls; red[1][wv] = lss; }
    __syncthreads();
    float S = 0.f, SS = 0.f;
    for (int i = 0; i < 8; ++i) { S += red[0][i]; SS += red[1][i]; }
    float mean = S / DD, var = SS / DD - mean * mean;
    float rstd = rsqrtf(var + 1e-5f);
    for (int d = threadIdx.x; d < DD; d += 256) {
        float v = (y[d] - mean) * rstd * pmg[d] + pmbn[d];
        float g = 0.5f * v * (1.0f + erff(v * 0.70710678118654752f));  // exact gelu
        float sig = expf(pls[t * DD + d]);
        for (int p = 0; p < PP; ++p)
            q[(((size_t)(b * TT + t) * PP) + p) * DD + d] = g + sig * rmul[((size_t)t * PP + p) * DD + d];
    }
}

// =====================================================================
// Kernel 5: K/V projection GEMM via WMMA bf16 (double-buffered pipeline)
//   Out[row, col] = sum_k x[row,k] * W[k,col] + bias[col]  (stored bf16)
//   block tile 64 rows x 256 cols, K stepped by 32; grid.z picks K vs V
//   8 waves: wave = mt (4 M-subtiles of 16) x nh (2 col halves of 128)
//   pipeline: global loads (k+1) -> regs overlap WMMA on LDS[cur];
//   one barrier per K-step; B-frags batched 4 at a time to clause ds_loads
// =====================================================================
__global__ __launch_bounds__(256) void pg_kv(const float* __restrict__ x,
                                             const float* __restrict__ Wk,
                                             const float* __restrict__ bk,
                                             const float* __restrict__ Wv,
                                             const float* __restrict__ bv,
                                             unsigned short* __restrict__ Kb_,
                                             unsigned short* __restrict__ Vb_) {
    __shared__ __bf16 As[2][64][32];    // 2 x 4 KB
    __shared__ __bf16 Bt[2][256][32];   // 2 x 16 KB, B tile transposed [n][k]
    const int tid = threadIdx.x;
    const int lane = tid & 31, wv = tid >> 5;
    const int mt = wv & 3, nh = wv >> 2;
    const size_t row0 = (size_t)blockIdx.x * 64;
    const int col0 = blockIdx.y * 256;
    const float* W = blockIdx.z ? Wv : Wk;
    const float* bias = blockIdx.z ? bv : bk;
    __bf16* Out = (__bf16*)(blockIdx.z ? Vb_ : Kb_);

    const int am = tid >> 2, aseg = tid & 3;
    const float* asrc = x + (row0 + am) * FF + aseg * 8;   // + k0
    const float* bsrc = W + col0 + tid;                    // + (k0+k)*DD

    float areg[8];
    float breg[32];
    // ---- prologue: stage tile k0 = 0 into LDS buffer 0 ----
#pragma unroll
    for (int e = 0; e < 8; ++e) areg[e] = asrc[e];
#pragma unroll
    for (int k = 0; k < 32; ++k) breg[k] = bsrc[(size_t)k * DD];
#pragma unroll
    for (int e = 0; e < 8; ++e) As[0][am][aseg * 8 + e] = (__bf16)areg[e];
#pragma unroll
    for (int k = 0; k < 32; ++k) Bt[0][tid][k] = (__bf16)breg[k];
    __syncthreads();

    v8f acc[8] = {};
    int cur = 0;
    const int mrow = mt * 16 + (lane & 15);
    const int kb = (lane >> 4) * 8;

    for (int k0 = 0; k0 < FF; k0 += 32) {
        const bool more = (k0 + 32) < FF;
        // ---- issue global loads for next tile (overlap with WMMA below) ----
        if (more) {
            const float* an = asrc + k0 + 32;
            const float* bn = bsrc + (size_t)(k0 + 32) * DD;
#pragma unroll
            for (int e = 0; e < 8; ++e) areg[e] = an[e];
#pragma unroll
            for (int k = 0; k < 32; ++k) breg[k] = bn[(size_t)k * DD];
            if (k0 + 64 < FF) {   // warm L2 for the tile after next (global_prefetch_b8)
                __builtin_prefetch(asrc + k0 + 64, 0, 1);
                __builtin_prefetch(bsrc + (size_t)(k0 + 64) * DD, 0, 1);
            }
        }
        // ---- A fragment (16x32, ISA 16-bit A layout) ----
        v16bf afrag;
        {
            const __bf16* p0 = &As[cur][mrow][kb];
            const __bf16* p1 = &As[cur][mrow][16 + kb];
#pragma unroll
            for (int e = 0; e < 8; ++e) { afrag[e] = p0[e]; afrag[8 + e] = p1[e]; }
        }
        // ---- B fragments in two batches of 4, then WMMA burst ----
#pragma unroll
        for (int g = 0; g < 2; ++g) {
            v16bf bfrag[4];
#pragma unroll
            for (int j = 0; j < 4; ++j) {
                int n = nh * 128 + (g * 4 + j) * 16 + (lane & 15);
                const __bf16* q0 = &Bt[cur][n][kb];
                const __bf16* q1 = &Bt[cur][n][16 + kb];
#pragma unroll
                for (int e = 0; e < 8; ++e) { bfrag[j][e] = q0[e]; bfrag[j][8 + e] = q1[e]; }
            }
#pragma unroll
            for (int j = 0; j < 4; ++j)
                acc[g * 4 + j] = __builtin_amdgcn_wmma_f32_16x16x32_bf16(
                    false, afrag, false, bfrag[j], (short)0, acc[g * 4 + j], false, false);
        }
        // ---- stage next tile into the other buffer, single barrier ----
        if (more) {
            int nxt = cur ^ 1;
#pragma unroll
            for (int e = 0; e < 8; ++e) As[nxt][am][aseg * 8 + e] = (__bf16)areg[e];
#pragma unroll
            for (int k = 0; k < 32; ++k) Bt[nxt][tid][k] = (__bf16)breg[k];
            __syncthreads();
            cur = nxt;
        }
    }
    // ---- epilogue: C layout -> row = r + (lane>=16 ? 8 : 0), col = lane&15 ----
#pragma unroll
    for (int nt = 0; nt < 8; ++nt) {
        int col = col0 + nh * 128 + nt * 16 + (lane & 15);
        float bvv = bias[col];
#pragma unroll
        for (int r = 0; r < 8; ++r) {
            size_t row = row0 + mt * 16 + r + ((lane >> 4) * 8);
            Out[row * DD + col] = (__bf16)(acc[nt][r] + bvv);
        }
    }
}

// =====================================================================
// Kernel 6: LayerNorm (no affine) over dh=64 on K, in place, bf16
// one wave per (row, h)
// =====================================================================
__global__ __launch_bounds__(256) void pg_kln(unsigned short* __restrict__ Kb_) {
    int lane = threadIdx.x & 31, wv = threadIdx.x >> 5;
    int g = blockIdx.x * 8 + wv;                 // < BB*NN*HH
    __bf16* p = (__bf16*)Kb_ + (size_t)(g >> 3) * DD + (g & 7) * DH;
    float v0 = (float)p[lane], v1 = (float)p[lane + 32];
    float s = v0 + v1, ss = v0 * v0 + v1 * v1;
    for (int off = 16; off; off >>= 1) { s += __shfl_xor(s, off, 32); ss += __shfl_xor(ss, off, 32); }
    float m = s * (1.0f / 64.0f), var = ss * (1.0f / 64.0f) - m * m;
    float r = rsqrtf(var + 1e-5f);
    p[lane] = (__bf16)((v0 - m) * r);
    p[lane + 32] = (__bf16)((v1 - m) * r);
}

// =====================================================================
// Kernel 7: qh = ln((q@Wq+bq).reshape(P,H,dh)) ; one block per b
// =====================================================================
__global__ __launch_bounds__(256) void pg_qh(const float* __restrict__ q,
                                             const float* __restrict__ Wq,
                                             const float* __restrict__ bq,
                                             float* __restrict__ qh, int t) {
    int b = blockIdx.x;
    __shared__ float qs[PP * DD];
    __shared__ float raw[PP * DD];
    for (int i = threadIdx.x; i < PP * DD; i += 256)
        qs[i] = q[(size_t)(b * TT + t) * PP * DD + i];
    __syncthreads();
    for (int o = threadIdx.x; o < PP * DD; o += 256) {
        int p = o >> 9, d = o & 511;
        const float* qp = qs + p * DD;
        float acc = bq[d];
        for (int k = 0; k < DD; ++k) acc += qp[k] * Wq[(size_t)k * DD + d];
        raw[o] = acc;
    }
    __syncthreads();
    if (threadIdx.x < PP * HH) {
        int p = threadIdx.x >> 3, h = threadIdx.x & 7;
        float* r = raw + p * DD + h * DH;
        float s = 0.f, ss = 0.f;
        for (int i = 0; i < DH; ++i) { float v = r[i]; s += v; ss += v * v; }
        float m = s / DH, var = ss / DH - m * m, rs = rsqrtf(var + 1e-5f);
        float* o = qh + (size_t)(b * PP + p) * DD + h * DH;
        for (int i = 0; i < DH; ++i) o[i] = (r[i] - m) * rs;
    }
}

// =====================================================================
// Kernel 8: dots + masked softmax, one block per (b,h)
// =====================================================================
__global__ __launch_bounds__(256) void pg_dots(const float* __restrict__ qh,
                                               const unsigned short* __restrict__ Kb_,
                                               const int* __restrict__ bm,
                                               float* __restrict__ dots) {
    int b = blockIdx.x, h = blockIdx.y;
    int lane = threadIdx.x & 31, wv = threadIdx.x >> 5;
    __shared__ float qs[PP][DH];
    __shared__ float red[16];
    const __bf16* Kb = (const __bf16*)Kb_;
    for (int i = threadIdx.x; i < PP * DH; i += 256)
        qs[i >> 6][i & 63] = qh[(size_t)(b * PP + (i >> 6)) * DD + h * DH + (i & 63)];
    __syncthreads();
    for (int n = threadIdx.x; n < NN; n += 256) {
        const __bf16* kp = Kb + (size_t)(b * NN + n) * DD + h * DH;
        float acc[PP] = {};
        for (int i = 0; i < DH; ++i) {
            float kv = (float)kp[i];
#pragma unroll
            for (int p = 0; p < PP; ++p) acc[p] += qs[p][i] * kv;
        }
        bool msk = (bm[b * NN + n] == 0);
#pragma unroll
        for (int p = 0; p < PP; ++p)
            dots[(((size_t)(b * PP + p) * HH) + h) * NN + n] = msk ? -INFINITY : acc[p] * 0.125f;
    }
    __syncthreads();
    for (int p = 0; p < PP; ++p) {
        float* dp = dots + (((size_t)(b * PP + p) * HH) + h) * NN;
        float mx = -INFINITY;
        for (int n = threadIdx.x; n < NN; n += 256) mx = fmaxf(mx, dp[n]);
        for (int off = 16; off; off >>= 1) mx = fmaxf(mx, __shfl_xor(mx, off, 32));
        if (lane == 0) red[wv] = mx;
        __syncthreads();
        float gmx = red[0];
        for (int i = 1; i < 8; ++i) gmx = fmaxf(gmx, red[i]);
        float sm = 0.f;
        for (int n = threadIdx.x; n < NN; n += 256) { float e = expf(dp[n] - gmx); dp[n] = e; sm += e; }
        for (int off = 16; off; off >>= 1) sm += __shfl_xor(sm, off, 32);
        if (lane == 0) red[8 + wv] = sm;
        __syncthreads();
        float gs = 0.f;
        for (int i = 0; i < 8; ++i) gs += red[8 + i];
        float inv = 1.0f / gs;
        for (int n = threadIdx.x; n < NN; n += 256) dp[n] *= inv;
        __syncthreads();
    }
}

// =====================================================================
// Kernel 9: upd[p, h*64+d] = sum_n attn[p,h,n] * V[n,h,d] ; block per (b,h)
// =====================================================================
__global__ __launch_bounds__(256) void pg_upd(const float* __restrict__ dots,
                                              const unsigned short* __restrict__ Vb_,
                                              float* __restrict__ upd) {
    int b = blockIdx.x, h = blockIdx.y;
    const __bf16* Vb = (const __bf16*)Vb_;
    __shared__ float at[PP][128];
    int p1 = threadIdx.x >> 6, d = threadIdx.x & 63;
    float a1 = 0.f, a2 = 0.f;
    for (int n0 = 0; n0 < NN; n0 += 128) {
        __syncthreads();
        for (int i = threadIdx.x; i < PP * 128; i += 256) {
            int p = i >> 7, nn = i & 127;
            at[p][nn] = dots[(((size_t)(b * PP + p) * HH) + h) * NN + n0 + nn];
        }
        __syncthreads();
        for (int nn = 0; nn < 128; ++nn) {
            float v = (float)Vb[(size_t)(b * NN + n0 + nn) * DD + h * DH + d];
            a1 += at[p1][nn] * v;
            a2 += at[p1 + 4][nn] * v;
        }
    }
    upd[(size_t)(b * PP + p1) * DD + h * DH + d] = a1;
    upd[(size_t)(b * PP + p1 + 4) * DD + h * DH + d] = a2;
}

// =====================================================================
// Kernel 10: layer tail — Wo proj + LN + resid, MLP (gelu) + LN + resid
// one block per b; 8 waves, wave-per-row LayerNorms
// =====================================================================
__device__ inline void pg_gemm8(const float* __restrict__ in, const float* __restrict__ Wg,
                                const float* __restrict__ bias, float* __restrict__ out) {
    for (int o = threadIdx.x; o < PP * DD; o += 256) {
        int p = o >> 9, d = o & 511;
        const float* ip = in + p * DD;
        float acc = bias[d];
        for (int k = 0; k < DD; ++k) acc += ip[k] * Wg[(size_t)k * DD + d];
        out[o] = acc;
    }
}

__global__ __launch_bounds__(256) void pg_tail(const float* __restrict__ upd,
                                               float* __restrict__ q,
                                               const float* Wo, const float* bo,
                                               const float* ag, const float* ab,
                                               const float* Wm1, const float* bm1,
                                               const float* lg, const float* lb,
                                               const float* Wm2, const float* bm2,
                                               const float* pg, const float* pb, int t) {
    __shared__ float A[PP * DD], Bs[PP * DD], Cs[PP * DD];   // 48 KB
    int b = blockIdx.x;
    int lane = threadIdx.x & 31, wv = threadIdx.x >> 5;
    float* qg = q + (size_t)(b * TT + t) * PP * DD;
    for (int i = threadIdx.x; i < PP * DD; i += 256) { Cs[i] = upd[(size_t)b * PP * DD + i]; A[i] = qg[i]; }
    __syncthreads();
    pg_gemm8(Cs, Wo, bo, Bs);                 // Bs = upd@Wo + bo
    __syncthreads();
    {   // A = A + ln(Bs; ag, ab)  (row-wise over 512, wave per row)
        const float* r = Bs + wv * DD;
        float s = 0.f, ss = 0.f;
        for (int dd = lane; dd < DD; dd += 32) { float v = r[dd]; s += v; ss += v * v; }
        for (int off = 16; off; off >>= 1) { s += __shfl_xor(s, off, 32); ss += __shfl_xor(ss, off, 32); }
        float m = s / DD, var = ss / DD - m * m, rs = rsqrtf(var + 1e-5f);
        for (int dd = lane; dd < DD; dd += 32)
            A[wv * DD + dd] += (r[dd] - m) * rs * ag[dd] + ab[dd];
    }
    __syncthreads();
    pg_gemm8(A, Wm1, bm1, Bs);                // Bs = q1@Wm1 + bm1
    __syncthreads();
    {   // Bs = gelu(ln(Bs; lg, lb))
        float* r = Bs + wv * DD;
        float s = 0.f, ss = 0.f;
        for (int dd = lane; dd < DD; dd += 32) { float v = r[dd]; s += v; ss += v * v; }
        for (int off = 16; off; off >>= 1) { s += __shfl_xor(s, off, 32); ss += __shfl_xor(ss, off, 32); }
        float m = s / DD, var = ss / DD - m * m, rs = rsqrtf(var + 1e-5f);
        for (int dd = lane; dd < DD; dd += 32) {
            float v = (r[dd] - m) * rs * lg[dd] + lb[dd];
            r[dd] = 0.5f * v * (1.0f + erff(v * 0.70710678118654752f));
        }
    }
    __syncthreads();
    pg_gemm8(Bs, Wm2, bm2, Cs);               // Cs = h@Wm2 + bm2
    __syncthreads();
    {   // qg = A + ln(Cs; pg, pb)
        const float* r = Cs + wv * DD;
        float s = 0.f, ss = 0.f;
        for (int dd = lane; dd < DD; dd += 32) { float v = r[dd]; s += v; ss += v * v; }
        for (int off = 16; off; off >>= 1) { s += __shfl_xor(s, off, 32); ss += __shfl_xor(ss, off, 32); }
        float m = s / DD, var = ss / DD - m * m, rs = rsqrtf(var + 1e-5f);
        for (int dd = lane; dd < DD; dd += 32)
            qg[wv * DD + dd] = A[wv * DD + dd] + (r[dd] - m) * rs * pg[dd] + pb[dd];
    }
}

// =====================================================================
// Kernel 11: copy final q state -> prototypes output
// =====================================================================
__global__ void pg_copy(const float* __restrict__ q, float* __restrict__ out) {
    int i = blockIdx.x * 256 + threadIdx.x;   // grid covers BB*TT*PP*DD
    out[i] = q[i];
}

// =====================================================================
// host launcher
// =====================================================================
extern "C" void kernel_launch(void* const* d_in, const int* in_sizes, int n_in,
                              void* d_out, int out_size, void* d_ws, size_t ws_size,
                              hipStream_t stream) {
    const float* x     = (const float*)d_in[0];
    const int*   bm    = (const int*)d_in[1];
    const int*   ntype = (const int*)d_in[2];
    const float* pmW   = (const float*)d_in[3];
    const float* pmb   = (const float*)d_in[4];
    const float* pmg   = (const float*)d_in[5];
    const float* pmbn  = (const float*)d_in[6];
    const float* pls   = (const float*)d_in[7];
    const float* rmul  = (const float*)d_in[8];
    const float* Wq = (const float*)d_in[9],  *bq = (const float*)d_in[10];
    const float* Wk = (const float*)d_in[11], *bk = (const float*)d_in[12];
    const float* Wv = (const float*)d_in[13], *bv = (const float*)d_in[14];
    const float* Wo = (const float*)d_in[15], *bo = (const float*)d_in[16];
    const float* ag = (const float*)d_in[17], *ab = (const float*)d_in[18];
    const float* Wm1 = (const float*)d_in[19], *bm1 = (const float*)d_in[20];
    const float* lg = (const float*)d_in[21], *lb = (const float*)d_in[22];
    const float* Wm2 = (const float*)d_in[23], *bm2 = (const float*)d_in[24];
    const float* pg = (const float*)d_in[25], *pb = (const float*)d_in[26];

    char* w = (char*)d_ws;
    size_t off = 0;
    auto take = [&](size_t bytes) -> char* {
        char* r = w + off;
        off = (off + bytes + 255) & ~(size_t)255;
        return r;
    };
    unsigned short* Kb = (unsigned short*)take((size_t)BB * NN * DD * 2);
    unsigned short* Vb = (unsigned short*)take((size_t)BB * NN * DD * 2);
    float* qst       = (float*)take((size_t)BB * TT * PP * DD * 4);
    float* qh        = (float*)take((size_t)BB * PP * DD * 4);
    float* dots      = (float*)take((size_t)BB * PP * HH * NN * 4);
    float* sums      = (float*)take((size_t)BB * TT * FF * 4);
    float* sums_part = (float*)take((size_t)NSPLIT * BB * TT * FF * 4);
    float* fall      = (float*)take((size_t)BB * FF * 4);
    float* fall_part = (float*)take((size_t)NSPLIT * BB * FF * 4);
    float* cnt       = (float*)take(256);
    float* bmsum     = (float*)take(256);
    float* valid     = (float*)take((size_t)BB * NN * 4);
    float* upd       = (float*)take((size_t)BB * PP * DD * 4);

    float* out_proto = (float*)d_out;                          // (B,T,P,D)
    float* out_pmu   = (float*)d_out + BB * TT * PP * DD;      // (B,T,F)

    // ---- phase 1: proto means ----
    pg_zero<<<1, 32, 0, stream>>>(cnt, bmsum);
    pg_valid<<<dim3(BB * NN / 8), 256, 0, stream>>>(x, bm, ntype, valid, cnt, bmsum);
    pg_sums<<<dim3(FF / 256, BB, NSPLIT), 256, 0, stream>>>(x, valid, ntype, sums_part, fall_part);
    pg_reduce<<<dim3(112), 256, 0, stream>>>(sums_part, fall_part, sums, fall);
    pg_mu<<<dim3(BB * TT), 256, 0, stream>>>(sums, cnt, fall, bmsum, pmW, pmb, pmg, pmbn,
                                             pls, rmul, qst, out_pmu);

    // ---- phase 2: 12 layer applications (types independent, iters sequential) ----
    for (int c = 0; c < LL; ++c) {
        int t = c >> 1;
        const float* Wq_c = Wq + (size_t)c * DD * DD; const float* bq_c = bq + (size_t)c * DD;
        const float* Wk_c = Wk + (size_t)c * FF * DD; const float* bk_c = bk + (size_t)c * DD;
        const float* Wv_c = Wv + (size_t)c * FF * DD; const float* bv_c = bv + (size_t)c * DD;
        const float* Wo_c = Wo + (size_t)c * DD * DD; const float* bo_c = bo + (size_t)c * DD;
        const float* ag_c = ag + (size_t)c * DD;      const float* ab_c = ab + (size_t)c * DD;
        const float* Wm1_c = Wm1 + (size_t)c * DD * DD; const float* bm1_c = bm1 + (size_t)c * DD;
        const float* lg_c = lg + (size_t)c * DD;      const float* lb_c = lb + (size_t)c * DD;
        const float* Wm2_c = Wm2 + (size_t)c * DD * DD; const float* bm2_c = bm2 + (size_t)c * DD;
        const float* pg_c = pg + (size_t)c * DD;      const float* pb_c = pb + (size_t)c * DD;

        pg_kv<<<dim3(BB * NN / 64, DD / 256, 2), 256, 0, stream>>>(x, Wk_c, bk_c, Wv_c, bv_c, Kb, Vb);
        pg_kln<<<dim3(BB * NN * HH / 8), 256, 0, stream>>>(Kb);
        pg_qh<<<dim3(BB), 256, 0, stream>>>(qst, Wq_c, bq_c, qh, t);
        pg_dots<<<dim3(BB, HH), 256, 0, stream>>>(qh, Kb, bm, dots);
        pg_upd<<<dim3(BB, HH), 256, 0, stream>>>(dots, Vb, upd);
        pg_tail<<<dim3(BB), 256, 0, stream>>>(upd, qst, Wo_c, bo_c, ag_c, ab_c,
                                              Wm1_c, bm1_c, lg_c, lb_c,
                                              Wm2_c, bm2_c, pg_c, pb_c, t);
    }

    // ---- phase 3: emit prototypes ----
    pg_copy<<<dim3(BB * TT * PP * DD / 256), 256, 0, stream>>>(qst, out_proto);
}
